// RNN_22170621182256
// MI455X (gfx1250) — matrix operand order repaired and checked by reading, not compiled, not run on hardware
//
#include <hip/hip_runtime.h>
#include <cmath>

// ---------------------------------------------------------------------------
// 2-layer tanh RNN for MI455X (gfx1250, wave32, WMMA + TDM).
//   Phase A (per layer): x_proj = X @ Wx^T + bx   -- bf16 WMMA GEMM,
//       B-panel (128KB) staged to LDS via tensor_load_to_lds (TDM),
//       ping-pong software pipeline (no cross-iteration register copies).
//   Phase B (per layer): h_t = tanh(x_proj_t + h_{t-1} @ Wh^T + bh)
//       16 persistent workgroups, Wh slice TDM-staged into LDS (128KB),
//       h state double-buffered in a 128KB L2-resident global buffer,
//       one device-wide sync per timestep; 4 independent WMMA accumulator
//       chains + ping-pong pipelined fragment loads.
// ---------------------------------------------------------------------------

#define B_ 32
#define T_ 1024
#define H_ 1024
#define L_ 2
#define M_ (B_ * T_)       // 32768 rows for the x-projection GEMM
#define KT_ (H_ >> 5)      // 32 k-tiles of depth 32

typedef __attribute__((ext_vector_type(16))) __bf16 v16bf;
typedef __attribute__((ext_vector_type(8)))  float  v8f;
typedef __attribute__((ext_vector_type(4)))  unsigned v4u;
typedef __attribute__((ext_vector_type(8)))  int      v8i;
typedef __attribute__((ext_vector_type(4)))  int      v4i;

// ---------------------------------------------------------------------------
// TDM: DMA a contiguous 128KB region (65536 bf16) from global into LDS.
// Descriptor per CDNA5 ISA: 2-D tile, 8 rows x 8192
// elements, data_size=1 (2B), no padding/iterate/multicast, type=2 ("image").
// Groups 2/3 (and the trailing group) zero for a 2-D tensor. Issued by a
// single wave; EXEC is ignored by TDM. Completion tracked with TENSORcnt.
// ---------------------------------------------------------------------------
__device__ __forceinline__ void tdm_load_contig_128k(const void* gsrc, void* lds_dst) {
  const unsigned lds_off = (unsigned)(uintptr_t)lds_dst;   // LDS aperture: low 32 bits = LDS byte offset
  const unsigned long long ga = (unsigned long long)(uintptr_t)gsrc;
  v4u g0;
  g0.x = 1u;                                            // count=1 (valid), user descriptor
  g0.y = lds_off;                                       // lds_addr [63:32]
  g0.z = (unsigned)(ga & 0xFFFFFFFFu);                  // global_addr [95:64]
  g0.w = (unsigned)((ga >> 32) & 0x01FFFFFFu)           // global_addr [120:96]
       | 0x80000000u;                                   // type=2 at [127:126]
  v8i g1;
  g1[0] = 0x00010000;          // workgroup_mask=0, data_size=1 (2 bytes)
  g1[1] = (int)(0x2000u << 16);// tensor_dim0 = 8192 (low 16 bits at [63:48])
  g1[2] = (int)(8u << 16);     // tensor_dim0 hi=0, tensor_dim1 = 8
  g1[3] = (int)(0x2000u << 16);// tensor_dim1 hi=0, tile_dim0 = 8192
  g1[4] = 8;                   // tile_dim1 = 8, tile_dim2 = 0
  g1[5] = 8192;                // tensor_dim0_stride = 8192 (low 32)
  g1[6] = 0;                   // stride0 hi, stride1 lo
  g1[7] = 0;                   // stride1 hi
  const v4i z4 = {0, 0, 0, 0};
  const v8i z8 = {0, 0, 0, 0, 0, 0, 0, 0};
  __builtin_amdgcn_tensor_load_to_lds(g0, g1, z4, z4, z8, 0);
  __builtin_amdgcn_s_wait_tensorcnt(0);
}

// ---------------------------------------------------------------------------
// Fragment packing (wave32 WMMA VGPR layouts, CDNA5 ISA §7.12.2)
// A (16x32 bf16):  lane = (m&15) | (((koff>>3)&1)<<4),
//                  elem = (koff&7) | ((koff>>4)<<3)          koff = k&31
// B (32x16 bf16):  lane = (n&15) | (((koff>>3)&1)<<4), elem = (koff&7) | ((koff>>4)<<3)
// Tiles stored as 512 contiguous bf16 -> one 32B contiguous load per lane.
// ---------------------------------------------------------------------------
__global__ void pack_a_kernel(const float* __restrict__ src, __bf16* __restrict__ dst,
                              int M, int K) {
  size_t i = (size_t)blockIdx.x * blockDim.x + threadIdx.x;
  if (i >= (size_t)M * K) return;
  int m = (int)(i / K);
  int k = (int)(i % K);
  int koff = k & 31;
  int half = (koff >> 3) & 1;
  int e    = (koff & 7) | ((koff >> 4) << 3);
  int lane = (m & 15) | (half << 4);
  size_t tile = (size_t)(m >> 4) * (size_t)(K >> 5) + (size_t)(k >> 5);
  dst[tile * 512 + lane * 16 + e] = (__bf16)__builtin_nontemporal_load(src + i);
}

// W is row-major [N x K] (out_features x in_features); we need B = W^T (K x N).
__global__ void pack_b_kernel(const float* __restrict__ src, __bf16* __restrict__ dst,
                              int N, int K) {
  size_t i = (size_t)blockIdx.x * blockDim.x + threadIdx.x;
  if (i >= (size_t)N * K) return;
  int n = (int)(i / K);
  int k = (int)(i % K);
  int koff = k & 31;
  int lane = (n & 15) | (((koff >> 3) & 1) << 4);
  int e    = (koff & 7) | ((koff >> 4) << 3);
  size_t tile = (size_t)(n >> 4) * (size_t)(K >> 5) + (size_t)(k >> 5);
  dst[tile * 512 + lane * 16 + e] = (__bf16)__builtin_nontemporal_load(src + i);
}

// ---------------------------------------------------------------------------
// Phase A: out[m,n] = sum_k A[m,k] * W[n,k] + bias[n]
// Block = 256 threads = 8 waves; wave w -> 16x64 output tile (4 independent
// WMMA chains). B panel (4 n-tiles x full K = 128KB) TDM-staged into LDS,
// shared by all 8 waves. Ping-pong pipeline: stage loads write directly into
// the buffer consumed in the next stage -> no rotation copies, partial waits.
// Grid = (N/64, M/128).
// ---------------------------------------------------------------------------
__global__ void __launch_bounds__(256, 1)
xproj_gemm_kernel(const __bf16* __restrict__ pa,
                  const __bf16* __restrict__ pb,
                  const float*  __restrict__ bias,
                  float*        __restrict__ out,
                  int M, int K, int N) {
  extern __shared__ char smem_raw[];
  __bf16* ldsB = (__bf16*)smem_raw;            // [4 n-tiles][ktiles][512]

  const int lane = threadIdx.x & 31;
  const int wave = threadIdx.x >> 5;
  const int m0   = blockIdx.y * 128 + wave * 16;
  const int ng   = blockIdx.x;                 // 64-wide n group
  const int ktiles = K >> 5;

  if (wave == 0)
    tdm_load_contig_128k(pb + (size_t)(ng * 4) * ktiles * 512, ldsB);
  __syncthreads();

  const __bf16* parow = pa + (size_t)(m0 >> 4) * ktiles * 512 + lane * 16;

  v8f acc[4] = {};
  v16bf aP, bP[4], aQ, bQ[4];                  // ping (P) / pong (Q) buffers

  // prologue: fill P with k-tile 0
  aP = *(const v16bf*)(parow);
#pragma unroll
  for (int j = 0; j < 4; ++j)
    bP[j] = *(const v16bf*)(ldsB + ((size_t)j * ktiles) * 512 + lane * 16);

  for (int kt = 0; kt < ktiles; kt += 2) {
    const int k1 = (kt + 1 < ktiles) ? kt + 1 : kt;
    const int k2 = (kt + 2 < ktiles) ? kt + 2 : kt;
    // stage 1: load Q(k+1), compute with P(k)
    aQ = *(const v16bf*)(parow + (size_t)k1 * 512);
#pragma unroll
    for (int j = 0; j < 4; ++j)
      bQ[j] = *(const v16bf*)(ldsB + ((size_t)j * ktiles + k1) * 512 + lane * 16);
#pragma unroll
    for (int j = 0; j < 4; ++j)
      acc[j] = __builtin_amdgcn_wmma_f32_16x16x32_bf16(false, aP, false, bP[j],
                                                       (short)0, acc[j], false, false);
    // stage 2: load P(k+2), compute with Q(k+1)
    aP = *(const v16bf*)(parow + (size_t)k2 * 512);
#pragma unroll
    for (int j = 0; j < 4; ++j)
      bP[j] = *(const v16bf*)(ldsB + ((size_t)j * ktiles + k2) * 512 + lane * 16);
#pragma unroll
    for (int j = 0; j < 4; ++j)
      acc[j] = __builtin_amdgcn_wmma_f32_16x16x32_bf16(false, aQ, false, bQ[j],
                                                       (short)0, acc[j], false, false);
  }

  // C/D layout: vgpr r, lanes 0-15 -> M=m0+r, lanes 16-31 -> M=m0+8+r; N=n0+(lane&15)
  const int nlo  = lane & 15;
  const int hsel = lane >> 4;
#pragma unroll
  for (int j = 0; j < 4; ++j) {
    const int n  = ng * 64 + j * 16 + nlo;
    const float bv = bias[n];
#pragma unroll
    for (int r = 0; r < 8; ++r) {
      const int m = m0 + hsel * 8 + r;
      __builtin_nontemporal_store(acc[j][r] + bv, out + (size_t)m * N + n);
    }
  }
}

// ---------------------------------------------------------------------------
// Device-wide step barrier: release fence + atomic arrive/wait (+ HW cluster
// barrier: cheap path if dispatched as a 16-WG cluster, S_NOP otherwise),
// then acquire fence (agent scope) so h_t stores from other WGPs are visible.
// ---------------------------------------------------------------------------
__device__ __forceinline__ void grid_barrier(unsigned* ctr, unsigned target) {
  __builtin_amdgcn_fence(__ATOMIC_RELEASE, "agent");
  __syncthreads();
  if (threadIdx.x == 0) {
    __hip_atomic_fetch_add(ctr, 1u, __ATOMIC_RELEASE, __HIP_MEMORY_SCOPE_AGENT);
    while (__hip_atomic_load(ctr, __ATOMIC_ACQUIRE, __HIP_MEMORY_SCOPE_AGENT) < target) {
      __builtin_amdgcn_s_sleep(2);
    }
  }
  __syncthreads();
  __builtin_amdgcn_s_cluster_barrier();
  __builtin_amdgcn_fence(__ATOMIC_ACQUIRE, "agent");
}

__global__ void init_ctr_kernel(unsigned* ctr) { *ctr = 0u; }

// ---------------------------------------------------------------------------
// Phase B: persistent recurrence. 16 blocks x 256 threads.
// Block blk owns output columns [64*blk, 64*blk+64). Its Wh slice
// (H x 64, packed B fragments, 128KB) is TDM-staged into LDS. h state is
// packed-A bf16, double buffered: read buf (t&1), write buf ((t+1)&1).
// Wave w: m-tile mt = w&1 (batch rows 16mt..), n-tile ntl = w>>1.
// 32 WMMAs per wave per step (K = 1024): 4 independent accumulator chains
// with ping-pong pipelined fragment loads (kt += 8, two 4-WMMA stages).
// ---------------------------------------------------------------------------
__global__ void __launch_bounds__(256, 1)
rnn_recurrence_kernel(const float*  __restrict__ xproj,  // [B*T, H], row = b*T+t
                      const __bf16* __restrict__ pWh,    // packed B, [H x H]
                      const float*  __restrict__ bh,     // [H]
                      float*        __restrict__ out,    // [B*T, H]
                      float*        __restrict__ hfin,   // [B, H]
                      __bf16*       packedH,             // 2 x (B x H) packed A
                      unsigned*     ctr) {
  extern __shared__ char smem_raw[];
  __bf16* ldsWh = (__bf16*)smem_raw;

  const int lane = threadIdx.x & 31;
  const int wave = threadIdx.x >> 5;
  const int blk  = blockIdx.x;                 // 0..15

  // TDM-stage this block's Wh slice: packed tiles for n-tiles [4blk,4blk+4)
  // are contiguous (128KB).
  if (wave == 0)
    tdm_load_contig_128k(pWh + (size_t)(blk * 4) * KT_ * 512, ldsWh);

  // Zero h buffer 0 for this block's column slice (k-tiles 2blk, 2blk+1).
  for (int i = threadIdx.x; i < 2048; i += blockDim.x) {
    const int mt = i >> 10;
    const int kt = 2 * blk + ((i >> 9) & 1);
    packedH[((size_t)mt * KT_ + kt) * 512 + (i & 511)] = (__bf16)0.0f;
  }
  grid_barrier(ctr, 16u);

  const int mt   = wave & 1;
  const int ntl  = wave >> 1;
  const int nlo  = lane & 15;
  const int hsel = lane >> 4;
  const int n    = blk * 64 + ntl * 16 + nlo;
  const float bhv = bh[n];
  const __bf16* ldsB = ldsWh + (size_t)ntl * KT_ * 512 + lane * 16;
  const int sliceElems = 2 * KT_ * 512;        // one [B x H] packed-A buffer

  for (int t = 0; t < T_; ++t) {
    const __bf16* hSrc = packedH + (size_t)(t & 1) * sliceElems
                       + (size_t)mt * KT_ * 512 + lane * 16;
    __bf16* hDst = packedH + (size_t)((t + 1) & 1) * sliceElems;

    v8f acc0 = {}, acc1 = {}, acc2 = {}, acc3 = {};
    v16bf aP[4], bP[4], aQ[4], bQ[4];          // ping / pong (4 k-tiles each)

    // prologue: fill P with k-tiles 0..3
#pragma unroll
    for (int j = 0; j < 4; ++j) {
      aP[j] = *(const v16bf*)(hSrc + (size_t)j * 512);
      bP[j] = *(const v16bf*)(ldsB + (size_t)j * 512);
    }
    for (int kt = 0; kt < KT_; kt += 8) {      // kt = 0, 8, 16, 24
      const int k1 = kt + 4;                   // always < KT_
      const int k2 = (kt + 8 < KT_) ? kt + 8 : kt;
      // stage 1: load Q(k+4..k+7), compute with P(k..k+3)
#pragma unroll
      for (int j = 0; j < 4; ++j) {
        aQ[j] = *(const v16bf*)(hSrc + (size_t)(k1 + j) * 512);
        bQ[j] = *(const v16bf*)(ldsB + (size_t)(k1 + j) * 512);
      }
      acc0 = __builtin_amdgcn_wmma_f32_16x16x32_bf16(false, aP[0], false, bP[0], (short)0, acc0, false, false);
      acc1 = __builtin_amdgcn_wmma_f32_16x16x32_bf16(false, aP[1], false, bP[1], (short)0, acc1, false, false);
      acc2 = __builtin_amdgcn_wmma_f32_16x16x32_bf16(false, aP[2], false, bP[2], (short)0, acc2, false, false);
      acc3 = __builtin_amdgcn_wmma_f32_16x16x32_bf16(false, aP[3], false, bP[3], (short)0, acc3, false, false);
      // stage 2: load P(k+8..k+11), compute with Q(k+4..k+7)
#pragma unroll
      for (int j = 0; j < 4; ++j) {
        aP[j] = *(const v16bf*)(hSrc + (size_t)(k2 + j) * 512);
        bP[j] = *(const v16bf*)(ldsB + (size_t)(k2 + j) * 512);
      }
      acc0 = __builtin_amdgcn_wmma_f32_16x16x32_bf16(false, aQ[0], false, bQ[0], (short)0, acc0, false, false);
      acc1 = __builtin_amdgcn_wmma_f32_16x16x32_bf16(false, aQ[1], false, bQ[1], (short)0, acc1, false, false);
      acc2 = __builtin_amdgcn_wmma_f32_16x16x32_bf16(false, aQ[2], false, bQ[2], (short)0, acc2, false, false);
      acc3 = __builtin_amdgcn_wmma_f32_16x16x32_bf16(false, aQ[3], false, bQ[3], (short)0, acc3, false, false);
    }
    const v8f acc = (acc0 + acc1) + (acc2 + acc3);

#pragma unroll
    for (int r = 0; r < 8; ++r) {
      const int bi = mt * 16 + hsel * 8 + r;                // batch row
      const size_t row = (size_t)bi * T_ + t;
      const float v = tanhf(acc[r]
                            + __builtin_nontemporal_load(xproj + row * H_ + n)
                            + bhv);
      __builtin_nontemporal_store(v, out + row * H_ + n);
      if (t == T_ - 1) hfin[(size_t)bi * H_ + n] = v;
      // repack h element (row bi, K index n) into packed-A layout for next step
      const int koff = n & 31;
      const int half = (koff >> 3) & 1;
      const int e    = (koff & 7) | ((koff >> 4) << 3);
      const int lp   = (bi & 15) | (half << 4);
      const size_t tile = (size_t)(bi >> 4) * KT_ + (n >> 5);
      hDst[tile * 512 + lp * 16 + e] = (__bf16)v;
    }
    // Prefetch next timestep's x_proj element (cold stream) into cache.
    if (t + 1 < T_) {
      const int bi0 = mt * 16 + hsel * 8;
      __builtin_prefetch(xproj + ((size_t)bi0 * T_ + (t + 1)) * H_ + n, 0, 1);
    }
    grid_barrier(ctr, 16u * (unsigned)(t + 2));
  }
}

// ---------------------------------------------------------------------------
// Host-side orchestration
// ---------------------------------------------------------------------------
extern "C" void kernel_launch(void* const* d_in, const int* in_sizes, int n_in,
                              void* d_out, int out_size, void* d_ws, size_t ws_size,
                              hipStream_t stream) {
  const float* x  = (const float*)d_in[0];   // [B,T,H]
  const float* Wx = (const float*)d_in[1];   // [L,H,H]
  const float* Wh = (const float*)d_in[2];   // [L,H,H]
  const float* bx = (const float*)d_in[3];   // [L,H]
  const float* bh = (const float*)d_in[4];   // [L,H]
  float* out = (float*)d_out;                // outputs [B,T,H] ++ h_n [L,B,H]
  float* outputs = out;
  float* h_n     = out + (size_t)B_ * T_ * H_;

  // Workspace carve-up
  char* p = (char*)d_ws;
  __bf16* packedA  = (__bf16*)p;  p += (size_t)M_ * H_ * 2;       // 64 MB
  float*  xproj    = (float*)p;   p += (size_t)M_ * H_ * 4;       // 128 MB
  __bf16* packedWx = (__bf16*)p;  p += (size_t)H_ * H_ * 2;       // 2 MB
  __bf16* packedWh = (__bf16*)p;  p += (size_t)H_ * H_ * 2;       // 2 MB
  __bf16* packedH  = (__bf16*)p;  p += (size_t)2 * B_ * H_ * 2;   // 128 KB
  unsigned* ctr    = (unsigned*)p;

  const int PK_BLK = 256;
  const int gridPackA = (int)(((size_t)M_ * H_ + PK_BLK - 1) / PK_BLK);
  const int gridPackW = (int)(((size_t)H_ * H_ + PK_BLK - 1) / PK_BLK);
  const dim3 gemmGrid(H_ / 64, M_ / 128);
  const size_t panelLds = (size_t)4 * KT_ * 512 * 2;   // 128 KB

  for (int l = 0; l < L_; ++l) {
    const float* layer_in = (l == 0) ? x : outputs;  // layer0 output lives in d_out
    pack_a_kernel<<<gridPackA, PK_BLK, 0, stream>>>(layer_in, packedA, M_, H_);
    pack_b_kernel<<<gridPackW, PK_BLK, 0, stream>>>(Wx + (size_t)l * H_ * H_, packedWx, H_, H_);
    pack_b_kernel<<<gridPackW, PK_BLK, 0, stream>>>(Wh + (size_t)l * H_ * H_, packedWh, H_, H_);
    xproj_gemm_kernel<<<gemmGrid, 256, panelLds, stream>>>(packedA, packedWx,
                                                           bx + (size_t)l * H_, xproj,
                                                           M_, H_, H_);
    init_ctr_kernel<<<1, 1, 0, stream>>>(ctr);
    rnn_recurrence_kernel<<<16, 256, panelLds, stream>>>(xproj, packedWh,
                                                         bh + (size_t)l * H_,
                                                         outputs,
                                                         h_n + (size_t)l * B_ * H_,
                                                         packedH, ctr);
  }
}
